// Transformer_85066122264620
// MI455X (gfx1250) — compile-verified, matrix operand order checked by reading, not run on hardware
//
#include <hip/hip_runtime.h>
#include <math.h>

// ---------------------------------------------------------------------------
// Model constants (must match reference)
// ---------------------------------------------------------------------------
#define DMODEL 512
#define NHEAD  8
#define DHEAD  64
#define FFDIM  2048
#define VOCAB  32000
#define NLE    6
#define NLD    6
#define BATCH  4
#define SLEN   512
#define TLEN   512

typedef __attribute__((ext_vector_type(16))) __bf16 v16bf;
typedef __attribute__((ext_vector_type(4)))  __bf16 v4bf;
typedef __attribute__((ext_vector_type(8)))  float  v8f;
typedef __attribute__((ext_vector_type(4)))  float  v4f;

// ---------------------------------------------------------------------------
// Batched WMMA GEMM:  C = alpha * (A @ B) + bias, optional ReLU, optional B^T
//   A: [M,K] (row stride lda), B: [K,N] (row stride ldb) or [N,K] if transB
//   batch z decomposed as (b = z / Hdim, h = z % Hdim) with independent strides
//
// Tile: 64x64 per 128-thread block (4 waves), BK = 32.
// LDS tiles are stored PRE-SWIZZLED in WMMA fragment order so each lane's
// 16 bf16 fragment elements are 32 contiguous bytes (-> ds_load_b128 x2),
// eliminating the scalar ds_load_u16 + v_mov_b16 packing.
//
// Fragment K index for element e (ISA 7.12.2): kk = (e&7) + half*8 + (e>>3)*16
// Inverse (per element): half = (kk>>3)&1 ; e = (kk&7) | (((kk>>4)&1)<<3)
// Any aligned group of 4 consecutive kk shares `half` and maps to 4
// consecutive e -> staging can use 8-byte LDS stores.
//
// Requires: M % 64 == 0, N % 64 == 0, K % 32 == 0 (true for all shapes here).
// ---------------------------------------------------------------------------
#define BM 64
#define BN 64
#define BK 32

__global__ __launch_bounds__(128) void wmma_gemm_kernel(
    const float* __restrict__ A, const float* __restrict__ Bm,
    const float* __restrict__ bias, float* __restrict__ C,
    int M, int N, int K,
    long long lda, long long ldb, long long ldc,
    int Hdim,
    long long sAb, long long sAh, long long sBb, long long sBh,
    long long sCb, long long sCh,
    int transB, int relu, float alpha)
{
    // fragment-order layouts:
    //   As[row][half][e]  (row 0..63, 32 bf16 per row)
    //   Bs[col][half][e]  (col 0..63, 32 bf16 per col)
    __shared__ alignas(64) __bf16 As[BM * BK];
    __shared__ alignas(64) __bf16 Bs[BK * BN];

    const int tid   = threadIdx.x;
    const int lane  = tid & 31;
    const int wid   = tid >> 5;      // wave id 0..3 -> row stripe
    const int r16   = lane & 15;
    const int halfI = lane >> 4;     // 0: lanes 0-15, 1: lanes 16-31

    const int tileM = blockIdx.y * BM;
    const int tileN = blockIdx.x * BN;
    const int z  = blockIdx.z;
    const int bz = z / Hdim;
    const int hz = z % Hdim;

    const float* Ap = A  + (long long)bz * sAb + (long long)hz * sAh
                         + (long long)tileM * lda;
    const float* Bp = Bm + (long long)bz * sBb + (long long)hz * sBh
                         + (transB ? (long long)tileN * ldb : (long long)tileN);
    float*       Cp = C  + (long long)bz * sCb + (long long)hz * sCh;

    v8f acc[4] = {};

    for (int k0 = 0; k0 < K; k0 += BK) {
        // ---- stage A: 64 rows x 8 k-groups of 4; b128 load -> b64 LDS store
        for (int g = tid; g < BM * (BK / 4); g += 128) {
            const int r  = g >> 3;
            const int kg = (g & 7) << 2;
            const v4f f = *(const v4f*)&Ap[(long long)r * lda + (k0 + kg)];
            const int hg = (kg >> 3) & 1;
            const int e0 = (kg & 7) | (((kg >> 4) & 1) << 3);
            v4bf o; o.x = (__bf16)f.x; o.y = (__bf16)f.y;
                    o.z = (__bf16)f.z; o.w = (__bf16)f.w;
            *(v4bf*)&As[r * 32 + hg * 16 + e0] = o;
        }
        // ---- stage B: 64 cols x 8 k-groups of 4
        for (int g = tid; g < BN * (BK / 4); g += 128) {
            const int n  = g & 63;
            const int kg = (g >> 6) << 2;
            v4f f;
            if (transB) {
                f = *(const v4f*)&Bp[(long long)n * ldb + (k0 + kg)];
            } else {
                f.x = Bp[(long long)(k0 + kg + 0) * ldb + n];
                f.y = Bp[(long long)(k0 + kg + 1) * ldb + n];
                f.z = Bp[(long long)(k0 + kg + 2) * ldb + n];
                f.w = Bp[(long long)(k0 + kg + 3) * ldb + n];
            }
            const int hg = (kg >> 3) & 1;
            const int e0 = (kg & 7) | (((kg >> 4) & 1) << 3);
            v4bf o; o.x = (__bf16)f.x; o.y = (__bf16)f.y;
                    o.z = (__bf16)f.z; o.w = (__bf16)f.w;
            *(v4bf*)&Bs[n * 32 + hg * 16 + e0] = o;
        }
        __syncthreads();

        // ---- fragments: one 32-byte LDS load each (ds_load_b128 x2)
        const v16bf af = *(const v16bf*)&As[(wid * 16 + r16) * 32 + halfI * 16];
#pragma unroll
        for (int j = 0; j < 4; ++j) {
            const v16bf bfv =
                *(const v16bf*)&Bs[(j * 16 + r16) * 32 + halfI * 16];
            acc[j] = __builtin_amdgcn_wmma_f32_16x16x32_bf16(
                false, af, false, bfv, (short)0, acc[j], false, false);
        }
        __syncthreads();
    }

    // epilogue: C/D layout (8 VGPRs): VGPR v -> M = v + halfI*8, N = r16
#pragma unroll
    for (int j = 0; j < 4; ++j) {
        const int col = tileN + j * 16 + r16;
        const float bv = bias ? bias[col] : 0.0f;
#pragma unroll
        for (int vv = 0; vv < 8; ++vv) {
            int row = tileM + wid * 16 + vv + halfI * 8;
            float val = acc[j][vv] * alpha + bv;
            if (relu) val = fmaxf(val, 0.0f);
            Cp[(long long)row * ldc + col] = val;
        }
    }
}

// ---------------------------------------------------------------------------
// Masked softmax over last dim of scores[BH, Tq, Tk], in place.
// mask element for (b, q, k) at mask[b*mb + q*mq + k]; 0 => -1e9.
// ---------------------------------------------------------------------------
__global__ __launch_bounds__(256) void softmax_mask_kernel(
    float* __restrict__ scores, const int* __restrict__ mask,
    int Tq, int Tk, int Hdim, long long mb, long long mq)
{
    __shared__ float red[256];
    const int tid = threadIdx.x;
    const int q = blockIdx.x;
    const int z = blockIdx.y;
    const int b = z / Hdim;
    float* row = scores + ((long long)z * Tq + q) * (long long)Tk;

    float mx = -3.4e38f;
    for (int k = tid; k < Tk; k += 256) {
        int m = mask[(long long)b * mb + (long long)q * mq + k];
        float v = (m == 0) ? -1e9f : row[k];
        row[k] = v;
        mx = fmaxf(mx, v);
    }
    red[tid] = mx;
    __syncthreads();
    for (int s = 128; s > 0; s >>= 1) {
        if (tid < s) red[tid] = fmaxf(red[tid], red[tid + s]);
        __syncthreads();
    }
    mx = red[0];
    __syncthreads();

    float sum = 0.0f;
    for (int k = tid; k < Tk; k += 256) {
        float e = expf(row[k] - mx);
        row[k] = e;
        sum += e;
    }
    red[tid] = sum;
    __syncthreads();
    for (int s = 128; s > 0; s >>= 1) {
        if (tid < s) red[tid] += red[tid + s];
        __syncthreads();
    }
    const float inv = 1.0f / red[0];
    for (int k = tid; k < Tk; k += 256) row[k] *= inv;
}

// ---------------------------------------------------------------------------
// out = LayerNorm(x + res) * g + b ;  res may be null (plain LN). In-place ok.
// One 256-thread block per row of length Dd.
// ---------------------------------------------------------------------------
__global__ __launch_bounds__(256) void add_ln_kernel(
    const float* __restrict__ x, const float* __restrict__ res,
    const float* __restrict__ g, const float* __restrict__ bt,
    float* __restrict__ out, int Dd)
{
    __shared__ float r1[256];
    __shared__ float r2[256];
    const int tid = threadIdx.x;
    const long long row = blockIdx.x;
    const float* xr = x + row * Dd;
    const float* rr = res ? res + row * Dd : nullptr;

    float s = 0.0f, s2 = 0.0f;
    for (int d = tid; d < Dd; d += 256) {
        float v = xr[d] + (rr ? rr[d] : 0.0f);
        s += v; s2 += v * v;
    }
    r1[tid] = s; r2[tid] = s2;
    __syncthreads();
    for (int st = 128; st > 0; st >>= 1) {
        if (tid < st) { r1[tid] += r1[tid + st]; r2[tid] += r2[tid + st]; }
        __syncthreads();
    }
    const float mu  = r1[0] / (float)Dd;
    const float var = r2[0] / (float)Dd - mu * mu;
    const float rs  = rsqrtf(var + 1e-5f);
    for (int d = tid; d < Dd; d += 256) {
        float v = xr[d] + (rr ? rr[d] : 0.0f);
        out[row * Dd + d] = (v - mu) * rs * g[d] + bt[d];
    }
}

// ---------------------------------------------------------------------------
// out[row, d] = emb[tok[row], d] + PE(row % Tlen, d)
// ---------------------------------------------------------------------------
__global__ __launch_bounds__(256) void embed_pe_kernel(
    const int* __restrict__ tok, const float* __restrict__ emb,
    float* __restrict__ out, int Tlen, int Dd)
{
    const int row = blockIdx.x;
    const int t = row % Tlen;
    const int id = tok[row];
    const float lg = logf(10000.0f) / (float)Dd;
    for (int d = threadIdx.x; d < Dd; d += 256) {
        float freq = expf(-(float)(2 * (d / 2)) * lg);
        float ang = (float)t * freq;
        float pe = (d & 1) ? cosf(ang) : sinf(ang);
        out[(long long)row * Dd + d] = emb[(long long)id * Dd + d] + pe;
    }
}

// ---------------------------------------------------------------------------
// Host-side helpers
// ---------------------------------------------------------------------------
static void gemm(hipStream_t st, const float* A, const float* Bm, const float* bias,
                 float* C, int M, int N, int K,
                 long long lda, long long ldb, long long ldc,
                 int batches, int Hdim,
                 long long sAb, long long sAh, long long sBb, long long sBh,
                 long long sCb, long long sCh,
                 int transB, int relu, float alpha)
{
    dim3 grid((unsigned)(N / BN), (unsigned)(M / BM), (unsigned)batches);
    wmma_gemm_kernel<<<grid, 128, 0, st>>>(A, Bm, bias, C, M, N, K, lda, ldb, ldc,
                                           Hdim, sAb, sAh, sBb, sBh, sCb, sCh,
                                           transB, relu, alpha);
}

// One full attention block (proj Q/K/V, scores+softmax, ctx, output proj -> sub)
static void attention(hipStream_t st,
                      const float* qin, int Tq,        // [BATCH, Tq, D]
                      const float* kvin, int Tk,       // [BATCH, Tk, D]
                      const float* Wq, const float* bq,
                      const float* Wk, const float* bk,
                      const float* Wv, const float* bv,
                      const float* Wo, const float* bo,
                      const int* mask, long long mb, long long mq,
                      float* qb, float* kb, float* vb,
                      float* ctx, float* sub, float* scores)
{
    const long long D = DMODEL;
    // projections over flattened rows
    gemm(st, qin,  Wq, bq, qb, BATCH * Tq, DMODEL, DMODEL, D, D, D,
         1, 1, 0, 0, 0, 0, 0, 0, 0, 0, 1.0f);
    gemm(st, kvin, Wk, bk, kb, BATCH * Tk, DMODEL, DMODEL, D, D, D,
         1, 1, 0, 0, 0, 0, 0, 0, 0, 0, 1.0f);
    gemm(st, kvin, Wv, bv, vb, BATCH * Tk, DMODEL, DMODEL, D, D, D,
         1, 1, 0, 0, 0, 0, 0, 0, 0, 0, 1.0f);

    // scores[b,h] = (Qh @ Kh^T) / sqrt(dk)   (batched over B*H heads)
    gemm(st, qb, kb, nullptr, scores, Tq, Tk, DHEAD,
         D, D, (long long)Tk,
         BATCH * NHEAD, NHEAD,
         (long long)Tq * D, DHEAD,                 // A head strides
         (long long)Tk * D, DHEAD,                 // B head strides
         (long long)NHEAD * Tq * Tk, (long long)Tq * Tk,
         /*transB=*/1, 0, 0.125f);

    softmax_mask_kernel<<<dim3((unsigned)Tq, BATCH * NHEAD), 256, 0, st>>>(
        scores, mask, Tq, Tk, NHEAD, mb, mq);

    // ctx[b,h] = P @ Vh
    gemm(st, scores, vb, nullptr, ctx, Tq, DHEAD, Tk,
         (long long)Tk, D, D,
         BATCH * NHEAD, NHEAD,
         (long long)NHEAD * Tq * Tk, (long long)Tq * Tk,
         (long long)Tk * D, DHEAD,
         (long long)Tq * D, DHEAD,
         0, 0, 1.0f);

    // output projection
    gemm(st, ctx, Wo, bo, sub, BATCH * Tq, DMODEL, DMODEL, D, D, D,
         1, 1, 0, 0, 0, 0, 0, 0, 0, 0, 1.0f);
}

// ---------------------------------------------------------------------------
// Entry point
// ---------------------------------------------------------------------------
extern "C" void kernel_launch(void* const* d_in, const int* in_sizes, int n_in,
                              void* d_out, int out_size, void* d_ws, size_t ws_size,
                              hipStream_t stream)
{
    (void)in_sizes; (void)n_in; (void)out_size; (void)ws_size;

    const int*   source     = (const int*)d_in[0];
    const int*   target     = (const int*)d_in[1];
    const int*   src_mask   = (const int*)d_in[2];   // [B,1,1,S]
    const int*   tgt_mask   = (const int*)d_in[3];   // [1,1,T,T] causal
    const float* emb        = (const float*)d_in[4];
    const float* enc_qkvo_w = (const float*)d_in[5];
    const float* enc_qkvo_b = (const float*)d_in[6];
    const float* enc_ff_w1  = (const float*)d_in[7];
    const float* enc_ff_b1  = (const float*)d_in[8];
    const float* enc_ff_w2  = (const float*)d_in[9];
    const float* enc_ff_b2  = (const float*)d_in[10];
    const float* enc_ln_g   = (const float*)d_in[11];
    const float* enc_ln_b   = (const float*)d_in[12];
    const float* enc_norm_g = (const float*)d_in[13];
    const float* enc_norm_b = (const float*)d_in[14];
    const float* dec_qkvo_w = (const float*)d_in[15];
    const float* dec_qkvo_b = (const float*)d_in[16];
    const float* dec_ff_w1  = (const float*)d_in[17];
    const float* dec_ff_b1  = (const float*)d_in[18];
    const float* dec_ff_w2  = (const float*)d_in[19];
    const float* dec_ff_b2  = (const float*)d_in[20];
    const float* dec_ln_g   = (const float*)d_in[21];
    const float* dec_ln_b   = (const float*)d_in[22];
    const float* dec_norm_g = (const float*)d_in[23];
    const float* dec_norm_b = (const float*)d_in[24];
    const float* out_w      = (const float*)d_in[25];
    const float* out_b      = (const float*)d_in[26];

    const size_t ND = (size_t)BATCH * SLEN * DMODEL;      // 1 Mi floats
    float* ws = (float*)d_ws;
    size_t off = 0;
    float* xe     = ws + off; off += ND;
    float* qb     = ws + off; off += ND;
    float* kb     = ws + off; off += ND;
    float* vb     = ws + off; off += ND;
    float* ctx    = ws + off; off += ND;
    float* sub    = ws + off; off += ND;
    float* encout = ws + off; off += ND;
    float* yb     = ws + off; off += ND;
    float* scores = ws + off; off += (size_t)BATCH * NHEAD * TLEN * TLEN;
    float* ffnh   = ws + off; off += (size_t)BATCH * SLEN * FFDIM;

    const long long D = DMODEL;
    const int BS = BATCH * SLEN;
    const int BT = BATCH * TLEN;
    const size_t DD = (size_t)DMODEL * DMODEL;

    // ----- Encoder -----
    embed_pe_kernel<<<BS, 256, 0, stream>>>(source, emb, xe, SLEN, DMODEL);
    for (int l = 0; l < NLE; ++l) {
        const float* W = enc_qkvo_w + (size_t)l * 4 * DD;
        const float* Bi = enc_qkvo_b + (size_t)l * 4 * DMODEL;
        const float* lg = enc_ln_g + (size_t)l * 2 * DMODEL;
        const float* lb = enc_ln_b + (size_t)l * 2 * DMODEL;

        attention(stream, xe, SLEN, xe, SLEN,
                  W + 0 * DD, Bi + 0 * DMODEL,
                  W + 1 * DD, Bi + 1 * DMODEL,
                  W + 2 * DD, Bi + 2 * DMODEL,
                  W + 3 * DD, Bi + 3 * DMODEL,
                  src_mask, /*mb=*/SLEN, /*mq=*/0,
                  qb, kb, vb, ctx, sub, scores);
        add_ln_kernel<<<BS, 256, 0, stream>>>(xe, sub, lg, lb, xe, DMODEL);

        gemm(stream, xe, enc_ff_w1 + (size_t)l * DMODEL * FFDIM,
             enc_ff_b1 + (size_t)l * FFDIM, ffnh, BS, FFDIM, DMODEL,
             D, FFDIM, FFDIM, 1, 1, 0, 0, 0, 0, 0, 0, 0, /*relu=*/1, 1.0f);
        gemm(stream, ffnh, enc_ff_w2 + (size_t)l * FFDIM * DMODEL,
             enc_ff_b2 + (size_t)l * DMODEL, sub, BS, DMODEL, FFDIM,
             FFDIM, D, D, 1, 1, 0, 0, 0, 0, 0, 0, 0, 0, 1.0f);
        add_ln_kernel<<<BS, 256, 0, stream>>>(xe, sub, lg + DMODEL, lb + DMODEL,
                                              xe, DMODEL);
    }
    add_ln_kernel<<<BS, 256, 0, stream>>>(xe, nullptr, enc_norm_g, enc_norm_b,
                                          encout, DMODEL);

    // ----- Decoder -----
    embed_pe_kernel<<<BT, 256, 0, stream>>>(target, emb, yb, TLEN, DMODEL);
    for (int l = 0; l < NLD; ++l) {
        const float* W = dec_qkvo_w + (size_t)l * 8 * DD;
        const float* Bi = dec_qkvo_b + (size_t)l * 8 * DMODEL;
        const float* lg = dec_ln_g + (size_t)l * 3 * DMODEL;
        const float* lb = dec_ln_b + (size_t)l * 3 * DMODEL;

        // masked self-attention (causal)
        attention(stream, yb, TLEN, yb, TLEN,
                  W + 0 * DD, Bi + 0 * DMODEL,
                  W + 1 * DD, Bi + 1 * DMODEL,
                  W + 2 * DD, Bi + 2 * DMODEL,
                  W + 3 * DD, Bi + 3 * DMODEL,
                  tgt_mask, /*mb=*/0, /*mq=*/TLEN,
                  qb, kb, vb, ctx, sub, scores);
        add_ln_kernel<<<BT, 256, 0, stream>>>(yb, sub, lg, lb, yb, DMODEL);

        // cross-attention over encoder output
        attention(stream, yb, TLEN, encout, SLEN,
                  W + 4 * DD, Bi + 4 * DMODEL,
                  W + 5 * DD, Bi + 5 * DMODEL,
                  W + 6 * DD, Bi + 6 * DMODEL,
                  W + 7 * DD, Bi + 7 * DMODEL,
                  src_mask, /*mb=*/SLEN, /*mq=*/0,
                  qb, kb, vb, ctx, sub, scores);
        add_ln_kernel<<<BT, 256, 0, stream>>>(yb, sub, lg + DMODEL, lb + DMODEL,
                                              yb, DMODEL);

        // FFN
        gemm(stream, yb, dec_ff_w1 + (size_t)l * DMODEL * FFDIM,
             dec_ff_b1 + (size_t)l * FFDIM, ffnh, BT, FFDIM, DMODEL,
             D, FFDIM, FFDIM, 1, 1, 0, 0, 0, 0, 0, 0, 0, /*relu=*/1, 1.0f);
        gemm(stream, ffnh, dec_ff_w2 + (size_t)l * FFDIM * DMODEL,
             dec_ff_b2 + (size_t)l * DMODEL, sub, BT, DMODEL, FFDIM,
             FFDIM, D, D, 1, 1, 0, 0, 0, 0, 0, 0, 0, 0, 1.0f);
        add_ln_kernel<<<BT, 256, 0, stream>>>(yb, sub, lg + 2 * DMODEL,
                                              lb + 2 * DMODEL, yb, DMODEL);
    }
    add_ln_kernel<<<BT, 256, 0, stream>>>(yb, nullptr, dec_norm_g, dec_norm_b,
                                          sub, DMODEL);

    // ----- Final vocab projection: [B*T, D] @ [D, V] + b -> d_out -----
    gemm(stream, sub, out_w, out_b, (float*)d_out, BT, VOCAB, DMODEL,
         D, VOCAB, VOCAB, 1, 1, 0, 0, 0, 0, 0, 0, 0, 0, 1.0f);
}
